// OrbitalAttention_71021579206855
// MI455X (gfx1250) — matrix operand order repaired and checked by reading, not compiled, not run on hardware
//
#include <hip/hip_runtime.h>
#include <hip/hip_bf16.h>

typedef _Float16 v16h __attribute__((ext_vector_type(16)));
typedef _Float16 v8h  __attribute__((ext_vector_type(8)));
typedef float    v8f  __attribute__((ext_vector_type(8)));

#define H  128
#define NH 4
#define HD 32

// ---- workspace layout (in _Float16 elements) ----
#define FRAG_HALFS 512           // 32 lanes x 16 halfs per 32x16 B fragment
#define WIN_KT 4                 // in_proj:  K=128 -> 4 k-tiles
#define WIN_NT 24                //           N=384 -> 24 n-tiles
#define WOUT_KT 4
#define WOUT_NT 8
#define WH1_KT 8                 // h1: K=256
#define WH1_NT 8
#define WH2_KT 4
#define WH2_NT 1                 // N=3 padded to 16
#define WTOK_KT 1                // tokenizer: K=6 padded to 32 (r_w rows 0-2, v_w rows 3-5)
#define WTOK_NT 8
#define OFF_WIN  0
#define OFF_WOUT (OFF_WIN  + WIN_KT*WIN_NT*FRAG_HALFS)    // 49152
#define OFF_WH1  (OFF_WOUT + WOUT_KT*WOUT_NT*FRAG_HALFS)  // 65536
#define OFF_WH2  (OFF_WH1  + WH1_KT*WH1_NT*FRAG_HALFS)    // 98304
#define OFF_WTOK (OFF_WH2  + WH2_KT*WH2_NT*FRAG_HALFS)    // 100352
#define WS_HALFS (OFF_WTOK + WTOK_KT*WTOK_NT*FRAG_HALFS)  // 104448

#define H2_SCALE     256.0f
#define H2_INV_SCALE (1.0f/256.0f)

// ---------------- prep: fp32 weights -> fp16 WMMA B-fragments ----------------
// B fragment element at (lane, half h):  N = 16*nt + (lane&15),
//                                        K = 32*kt + 16*(lane>>4) + h
__device__ __forceinline__ void pack_one(const float* __restrict__ W, int Kdim,
                                         int Nvalid, int ntiles, float scale,
                                         _Float16* __restrict__ dst, int e) {
  int frag   = e >> 9;        // /512
  int within = e & 511;
  int lane   = within >> 4;
  int h      = within & 15;
  int kt     = frag / ntiles;
  int nt     = frag - kt * ntiles;
  int K = 32 * kt + 16 * (lane >> 4) + h;
  int N = 16 * nt + (lane & 15);
  float v = (N < Nvalid) ? W[N * Kdim + K] * scale : 0.0f;
  dst[e] = (_Float16)v;
}

__global__ void orbital_prep_kernel(const float* __restrict__ in_proj_w,
                                    const float* __restrict__ out_proj_w,
                                    const float* __restrict__ h1_w,
                                    const float* __restrict__ h2_w,
                                    const float* __restrict__ r_w,
                                    const float* __restrict__ v_w,
                                    _Float16* __restrict__ ws) {
  int tid = blockIdx.x * blockDim.x + threadIdx.x;
  if (tid >= WS_HALFS) return;
  if (tid < OFF_WOUT) {
    pack_one(in_proj_w, 128, 384, WIN_NT, 1.0f, ws + OFF_WIN, tid - OFF_WIN);
  } else if (tid < OFF_WH1) {
    pack_one(out_proj_w, 128, 128, WOUT_NT, 1.0f, ws + OFF_WOUT, tid - OFF_WOUT);
  } else if (tid < OFF_WH2) {
    pack_one(h1_w, 256, 128, WH1_NT, 1.0f, ws + OFF_WH1, tid - OFF_WH1);
  } else if (tid < OFF_WTOK) {
    pack_one(h2_w, 128, 3, WH2_NT, H2_SCALE, ws + OFF_WH2, tid - OFF_WH2);
  } else {
    // tokenizer fragment: K rows 0..2 = r_w[n][k], K rows 3..5 = v_w[n][k-3], else 0
    int e      = tid - OFF_WTOK;
    int frag   = e >> 9;
    int within = e & 511;
    int lane   = within >> 4;
    int h      = within & 15;
    int K = 16 * (lane >> 4) + h;          // kt == 0
    int N = 16 * frag + (lane & 15);
    float v = 0.0f;
    if (K < 3)      v = r_w[N * 3 + K];
    else if (K < 6) v = v_w[N * 3 + (K - 3)];
    ws[OFF_WTOK + e] = (_Float16)v;
  }
}

// ---------------- fragment helpers ----------------
// A fragment (16x32 f16, row-major source with `stride` halfs/row):
// lane l holds row rbase+(l&15); K runs [16*(l>>4), +16)
__device__ __forceinline__ v16h load_a_frag(const _Float16* buf, int rbase,
                                            int stride, int kbase, int lane) {
  const _Float16* p = buf + (rbase + (lane & 15)) * stride + kbase + 16 * (lane >> 4);
  v8h lo = *(const v8h*)p;
  v8h hi = *(const v8h*)(p + 8);
  return __builtin_shufflevector(lo, hi, 0,1,2,3,4,5,6,7,8,9,10,11,12,13,14,15);
}

__device__ __forceinline__ v16h load_b_frag(const _Float16* __restrict__ ws,
                                            int fragIdx, int lane) {
  return *(const v16h*)(ws + fragIdx * FRAG_HALFS + lane * 16);
}

#define WMMA(acc, a, b) \
  __builtin_amdgcn_wmma_f32_16x16x32_f16(false, (a), false, (b), (short)0, (acc), false, false)

// Wave-local LDS fence: waves are independent, so a full workgroup barrier is
// overkill. Lanes of one wave run in lockstep; LDS ops are in-order per wave,
// so an explicit split-counter wait (CDNA5 s_wait_dscnt) is sufficient.
__device__ __forceinline__ void wave_lds_fence() {
  __builtin_amdgcn_wave_barrier();
  asm volatile("s_wait_dscnt 0x0" ::: "memory");
  __builtin_amdgcn_wave_barrier();
}

// ---------------- main kernel: 4 waves/WG, 16 items/wave ----------------
__global__ __launch_bounds__(128)
void orbital_main_kernel(const float* __restrict__ state,
                         const float* __restrict__ r_b,
                         const float* __restrict__ v_b,
                         const float* __restrict__ pos_emb,
                         const float* __restrict__ in_proj_b,
                         const float* __restrict__ out_proj_b,
                         const float* __restrict__ h1_b,
                         const float* __restrict__ h2_b,
                         const _Float16* __restrict__ ws,
                         float* __restrict__ out) {
  extern __shared__ _Float16 smem[];
  const int lane = threadIdx.x & 31;
  const int wave = threadIdx.x >> 5;

  // per-wave LDS slab: 20480 halfs = 40 KB  (4 waves -> 160 KB -> 2 WG / WGP)
  _Float16* xbuf = smem + wave * 20480;  // 32 x 128 (token rows), also 16 x 256 flat
  _Float16* qkv  = xbuf + 4096;          // 32 x 384
  _Float16* ctx  = qkv + 12288;          // 32 x 128
  _Float16* hbuf = qkv;                  // 16 x 128, aliases qkv (dead after attention)

  const int itemBase = (blockIdx.x * 4 + wave) * 16;

  const _Float16* wIn  = ws + OFF_WIN;
  const _Float16* wOut = ws + OFF_WOUT;
  const _Float16* wH1  = ws + OFF_WH1;
  const _Float16* wH2  = ws + OFF_WH2;
  const _Float16* wTok = ws + OFF_WTOK;

  // ---- tokenizer as WMMA (M=32, K=6 padded to 32, N=128) : 16 WMMA ----
  // A: even rows carry state[0:3] in K 0..2, odd rows carry state[3:6] in K 3..5.
  // Nonzero K (<6) live entirely in lanes 0..15, halfs 0..5 of the A fragment.
  for (int mt = 0; mt < 2; ++mt) {
    v16h a;
#pragma unroll
    for (int i = 0; i < 16; ++i) a[i] = (_Float16)0.f;
    if (lane < 16) {
      const int item = itemBase + ((mt * 16 + lane) >> 1);
      const float* s = state + (size_t)item * 6;
      if ((lane & 1) == 0) {
        a[0] = (_Float16)s[0]; a[1] = (_Float16)s[1]; a[2] = (_Float16)s[2];
      } else {
        a[3] = (_Float16)s[3]; a[4] = (_Float16)s[4]; a[5] = (_Float16)s[5];
      }
    }
    for (int nt = 0; nt < WTOK_NT; ++nt) {
      const int col = nt * 16 + (lane & 15);
      const float b0 = r_b[col] + pos_emb[col];       // token 0 bias (even rows)
      const float b1 = v_b[col] + pos_emb[H + col];   // token 1 bias (odd rows)
      v8f acc;
#pragma unroll
      for (int r = 0; r < 8; ++r) acc[r] = (r & 1) ? b1 : b0;
      v16h b = load_b_frag(wTok, nt, lane);
      acc = WMMA(acc, a, b);
#pragma unroll
      for (int r = 0; r < 8; ++r) {
        const int row = mt * 16 + r + 8 * (lane >> 4);
        xbuf[row * H + col] = (_Float16)acc[r];
      }
    }
  }
  wave_lds_fence();

  // ---- in_proj: qkv = x @ Win^T + b   (M=32, K=128, N=384) : 192 WMMA ----
  for (int mt = 0; mt < 2; ++mt) {
    for (int nt = 0; nt < WIN_NT; ++nt) {
      const float bias = in_proj_b[nt * 16 + (lane & 15)];
      v8f acc; for (int i = 0; i < 8; ++i) acc[i] = bias;
#pragma unroll
      for (int kt = 0; kt < WIN_KT; ++kt) {
        v16h a = load_a_frag(xbuf, mt * 16, H, kt * 32, lane);
        v16h b = load_b_frag(wIn, kt * WIN_NT + nt, lane);
        acc = WMMA(acc, a, b);
      }
      const int col = nt * 16 + (lane & 15);
#pragma unroll
      for (int r = 0; r < 8; ++r) {
        const int row = mt * 16 + r + 8 * (lane >> 4);
        qkv[row * 384 + col] = (_Float16)acc[r];
      }
    }
  }
  wave_lds_fence();

  // ---- attention (seq len 2, 4 heads x 32): one lane per token row ----
  {
    const _Float16* qrow = qkv + lane * 384;
    const _Float16* k0 = qkv + (lane & ~1) * 384 + H;
    const _Float16* k1 = qkv + (lane |  1) * 384 + H;
    const _Float16* v0 = qkv + (lane & ~1) * 384 + 2 * H;
    const _Float16* v1 = qkv + (lane |  1) * 384 + 2 * H;
    const float scale = 0.17677669529663687f;  // 1/sqrt(32)
    for (int hh = 0; hh < NH; ++hh) {
      const int o = hh * HD;
      float s0 = 0.f, s1 = 0.f;
#pragma unroll
      for (int d = 0; d < HD; ++d) {
        const float qv = (float)qrow[o + d];
        s0 += qv * (float)k0[o + d];
        s1 += qv * (float)k1[o + d];
      }
      s0 *= scale; s1 *= scale;
      const float m  = fmaxf(s0, s1);
      const float e0 = __expf(s0 - m), e1 = __expf(s1 - m);
      const float inv = 1.f / (e0 + e1);
      const float a0 = e0 * inv, a1 = e1 * inv;
#pragma unroll
      for (int d = 0; d < HD; ++d) {
        const float cv = a0 * (float)v0[o + d] + a1 * (float)v1[o + d];
        ctx[lane * H + o + d] = (_Float16)cv;
      }
    }
  }
  wave_lds_fence();

  // ---- out_proj + residual into xbuf (M=32, K=128, N=128) : 64 WMMA ----
  for (int mt = 0; mt < 2; ++mt) {
    for (int nt = 0; nt < WOUT_NT; ++nt) {
      const float bias = out_proj_b[nt * 16 + (lane & 15)];
      v8f acc; for (int i = 0; i < 8; ++i) acc[i] = bias;
#pragma unroll
      for (int kt = 0; kt < WOUT_KT; ++kt) {
        v16h a = load_a_frag(ctx, mt * 16, H, kt * 32, lane);
        v16h b = load_b_frag(wOut, kt * WOUT_NT + nt, lane);
        acc = WMMA(acc, a, b);
      }
      const int col = nt * 16 + (lane & 15);
#pragma unroll
      for (int r = 0; r < 8; ++r) {
        const int row = mt * 16 + r + 8 * (lane >> 4);
        const float xv = (float)xbuf[row * H + col];
        xbuf[row * H + col] = (_Float16)(xv + acc[r]);   // residual, in place
      }
    }
  }
  wave_lds_fence();

  // ---- h1 + SiLU: xbuf viewed as 16 items x 256 (M=16, K=256, N=128) : 64 WMMA ----
  for (int nt = 0; nt < WH1_NT; ++nt) {
    const float bias = h1_b[nt * 16 + (lane & 15)];
    v8f acc; for (int i = 0; i < 8; ++i) acc[i] = bias;
#pragma unroll
    for (int kt = 0; kt < WH1_KT; ++kt) {
      v16h a = load_a_frag(xbuf, 0, 2 * H, kt * 32, lane);
      v16h b = load_b_frag(wH1, kt * WH1_NT + nt, lane);
      acc = WMMA(acc, a, b);
    }
    const int col = nt * 16 + (lane & 15);
#pragma unroll
    for (int r = 0; r < 8; ++r) {
      const int row = r + 8 * (lane >> 4);               // item index 0..15
      const float a = acc[r];
      const float sg = 1.f / (1.f + __expf(-a));
      hbuf[row * H + col] = (_Float16)(a * sg);          // hbuf aliases dead qkv
    }
  }
  wave_lds_fence();

  // ---- h2 (M=16, K=128, N=3 padded to 16, weights pre-scaled) : 4 WMMA ----
  {
    const int n = lane & 15;
    const float bias = (n < 3) ? h2_b[n] * H2_SCALE : 0.f;
    v8f acc; for (int i = 0; i < 8; ++i) acc[i] = bias;
#pragma unroll
    for (int kt = 0; kt < WH2_KT; ++kt) {
      v16h a = load_a_frag(hbuf, 0, H, kt * 32, lane);
      v16h b = load_b_frag(wH2, kt, lane);
      acc = WMMA(acc, a, b);
    }
    if (n < 3) {
#pragma unroll
      for (int r = 0; r < 8; ++r) {
        const int item = itemBase + r + 8 * (lane >> 4);
        out[(size_t)item * 3 + n] = acc[r] * H2_INV_SCALE;
      }
    }
  }
}

extern "C" void kernel_launch(void* const* d_in, const int* in_sizes, int n_in,
                              void* d_out, int out_size, void* d_ws, size_t ws_size,
                              hipStream_t stream) {
  const float* state      = (const float*)d_in[0];
  const float* r_w        = (const float*)d_in[1];
  const float* r_b        = (const float*)d_in[2];
  const float* v_w        = (const float*)d_in[3];
  const float* v_b        = (const float*)d_in[4];
  const float* pos_emb    = (const float*)d_in[5];
  const float* in_proj_w  = (const float*)d_in[6];
  const float* in_proj_b  = (const float*)d_in[7];
  const float* out_proj_w = (const float*)d_in[8];
  const float* out_proj_b = (const float*)d_in[9];
  const float* h1_w       = (const float*)d_in[10];
  const float* h1_b       = (const float*)d_in[11];
  const float* h2_w       = (const float*)d_in[12];
  const float* h2_b       = (const float*)d_in[13];
  _Float16* ws  = (_Float16*)d_ws;
  float*    out = (float*)d_out;

  // 1) pack all weights (incl. fused tokenizer r_w/v_w) into fp16 B-fragments
  orbital_prep_kernel<<<(WS_HALFS + 255) / 256, 256, 0, stream>>>(
      in_proj_w, out_proj_w, h1_w, h2_w, r_w, v_w, ws);

  // 2) main fused kernel: 64 items per 128-thread block (16 per wave)
  const int B = in_sizes[0] / 6;
  const int blocks = B / 64;
  const size_t shmem = (size_t)4 * 20480 * sizeof(_Float16);  // 160 KB -> 2 WG / WGP
  orbital_main_kernel<<<blocks, 128, shmem, stream>>>(
      state, r_b, v_b, pos_emb,
      in_proj_b, out_proj_b, h1_b, h2_b, ws, out);
}
